// ParallelGNNBlock_14353780703440
// MI455X (gfx1250) — matrix operand order repaired and checked by reading, compiled, not run on hardware
//
#include <hip/hip_runtime.h>

// ---------------- problem constants (from reference) ----------------
static constexpr int N     = 8192;
static constexpr int DIN   = 512;
static constexpr int DOUT  = 128;
static constexpr int HN    = 256;      // concat width of h = [h1 | h2]
static constexpr int E     = 131072;
static constexpr float ALPHA = 0.01f;

// ---------------- vector types for WMMA ----------------
typedef __attribute__((ext_vector_type(16))) __bf16 v16bf;
typedef __attribute__((ext_vector_type(8)))  __bf16 v8bf;
typedef __attribute__((ext_vector_type(8)))  float  v8f;
typedef __attribute__((ext_vector_type(4)))  int    v4i_t;

static __device__ inline v8f wmma_bf16(v16bf a, v16bf b, v8f c) {
    // (neg_a, A, neg_b, B, c_mod, C, reuse_a, reuse_b)
    return __builtin_amdgcn_wmma_f32_16x16x32_bf16(false, a, false, b, (short)0, c, false, false);
}

// ---------------- gfx1250 async global->LDS copy (guarded, with fallback) ----------
#if defined(__gfx1250__) && __has_builtin(__builtin_amdgcn_global_load_async_to_lds_b128)
#define USE_ASYNC_LDS 1
#else
#define USE_ASYNC_LDS 0
#endif

#define AS_GLOBAL __attribute__((address_space(1)))
#define AS_LDS    __attribute__((address_space(3)))

static __device__ inline void copy16_g2l(const __bf16* g, __bf16* l) {
#if USE_ASYNC_LDS
    __builtin_amdgcn_global_load_async_to_lds_b128(
        (AS_GLOBAL v4i_t*)g, (AS_LDS v4i_t*)l, 0, 0);
#else
    *reinterpret_cast<uint4*>(l) = *reinterpret_cast<const uint4*>(g);
#endif
}

static __device__ inline void async_fence() {
#if USE_ASYNC_LDS
#if __has_builtin(__builtin_amdgcn_s_wait_asynccnt)
    __builtin_amdgcn_s_wait_asynccnt(0);
#else
    asm volatile("s_wait_asynccnt 0x0" ::: "memory");
#endif
#endif
}

// ---------------- 0: init small arrays ----------------
__global__ void k_init(unsigned* degcnt, unsigned* counts) {
    int i = blockIdx.x * blockDim.x + threadIdx.x;
    if (i < N) { degcnt[i] = 0u; counts[i] = 0u; }
}

// ---------------- 1: convert x -> bf16 hi/lo ----------------
__global__ void k_conv_x(const float* __restrict__ x, __bf16* __restrict__ xhi,
                         __bf16* __restrict__ xlo) {
    int i = blockIdx.x * blockDim.x + threadIdx.x;   // N*DIN threads
    float v = x[i];
    __bf16 h = (__bf16)v;
    xhi[i] = h;
    xlo[i] = (__bf16)(v - (float)h);
}

// ---------------- 2: convert W1|W2 -> bf16 hi/lo, pre-swizzled into WMMA B layout --
// B element j of lane (r,half) in k-tile kk, n-tile nt corresponds to
//   Wc[k = kk*32 + (j < 8 ? 8*half + j : 16 + 8*half + (j - 8))][n = nt*16 + r]
// stored contiguously: swz[(((kk*16)+nt)*32 + lane)*16 + j]
__global__ void k_conv_w(const float* __restrict__ W1, const float* __restrict__ W2,
                         __bf16* __restrict__ whi, __bf16* __restrict__ wlo) {
    int id = blockIdx.x * blockDim.x + threadIdx.x;  // 131072 threads
    int j    = id & 15;
    int lane = (id >> 4) & 31;
    int nt   = (id >> 9) & 15;
    int kk   = id >> 13;
    int r = lane & 15, half = lane >> 4;
    int k = kk * 32 + (j < 8 ? 8 * half + j : 16 + 8 * half + (j - 8));
    int n = nt * 16 + r;
    float v = (n < DOUT) ? W1[k * DOUT + n] : W2[k * DOUT + (n - DOUT)];
    __bf16 h = (__bf16)v;
    whi[id] = h;
    wlo[id] = (__bf16)(v - (float)h);
}

// ---------------- 3: split-bf16 WMMA GEMM with LDS-staged A tile -------------------
// Block = 8 waves; all waves share one M-tile (16 rows of x, hi+lo = 32KB) staged
// into LDS once (async-to-LDS when available), each wave owns one N-tile.
static constexpr int APITCH = 520;   // 512 + 8 bf16: 1040B row pitch => ds_load_b128 bank-conflict-free

__global__ __launch_bounds__(256)
void k_gemm(const __bf16* __restrict__ xhi, const __bf16* __restrict__ xlo,
            const __bf16* __restrict__ whi, const __bf16* __restrict__ wlo,
            float* __restrict__ h) {
    __shared__ __bf16 sA[2 * 16 * APITCH];           // [hi|lo][16 rows][520]
    int tid  = threadIdx.x;
    int wave = tid >> 5, lane = tid & 31;
    int mt = blockIdx.x >> 1;                        // 512 row tiles, 2 blocks each
    int nt = (blockIdx.x & 1) * 8 + wave;            // 16 col tiles
    int r = lane & 15, half = lane >> 4;

    const __bf16* gHi = xhi + (size_t)mt * 16 * DIN; // contiguous 16KB
    const __bf16* gLo = xlo + (size_t)mt * 16 * DIN;
    __bf16* lHi = sA;
    __bf16* lLo = sA + 16 * APITCH;

    // stage A: 2 buffers x 1024 x 16B chunks, 8 chunks per thread
    for (int q = 0; q < 8; ++q) {
        int c = q * 256 + tid;                       // 0..2047
        int cc = c & 1023;
        int row = cc >> 6, col8 = cc & 63;
        const __bf16* gsrc = ((c >> 10) ? gLo : gHi) + row * DIN + col8 * 8;
        __bf16*       ldst = ((c >> 10) ? lLo : lHi) + row * APITCH + col8 * 8;
        copy16_g2l(gsrc, ldst);
    }
    async_fence();
    __syncthreads();

    v8f acc = {};
    const __bf16* lah = lHi + r * APITCH + 8 * half;
    const __bf16* lal = lLo + r * APITCH + 8 * half;

    for (int kk = 0; kk < 16; ++kk) {
        int ko = kk * 32;
        union { v16bf v; v8bf c[2]; } ahi, alo;
        ahi.c[0] = *reinterpret_cast<const v8bf*>(lah + ko);
        ahi.c[1] = *reinterpret_cast<const v8bf*>(lah + ko + 16);
        alo.c[0] = *reinterpret_cast<const v8bf*>(lal + ko);
        alo.c[1] = *reinterpret_cast<const v8bf*>(lal + ko + 16);

        size_t bofs = (size_t)(((kk * 16) + nt) * 32 + lane) * 16;
        v16bf bhi = *reinterpret_cast<const v16bf*>(whi + bofs);
        v16bf blo = *reinterpret_cast<const v16bf*>(wlo + bofs);

        acc = wmma_bf16(ahi.v, bhi, acc);   // hi*hi
        acc = wmma_bf16(ahi.v, blo, acc);   // hi*lo
        acc = wmma_bf16(alo.v, bhi, acc);   // lo*hi
    }

    // C/D layout: VGPR i -> M = i + 8*half, N = r
    size_t base = (size_t)(mt * 16 + 8 * half) * HN + nt * 16 + r;
    for (int i = 0; i < 8; ++i) h[base + (size_t)i * HN] = acc[i];
}

// ---------------- 4: per-edge counting (out-degree by src, in-degree by dst) -------
__global__ void k_edge_count(const int* __restrict__ ei, unsigned* __restrict__ counts,
                             unsigned* __restrict__ degcnt) {
    int e = blockIdx.x * blockDim.x + threadIdx.x;
    if (e < E) {
        atomicAdd(&counts[ei[e]], 1u);       // src (out-CSR)
        atomicAdd(&degcnt[ei[E + e]], 1u);   // dst (in-CSR + deg1)
    }
}

// ---------------- 5: single-block exclusive scan: cnt -> offsets, fillptr ----------
__global__ __launch_bounds__(256)
void k_scan(const unsigned* __restrict__ cnt, unsigned* __restrict__ offsets,
            unsigned* __restrict__ fillptr) {
    __shared__ unsigned part[256];
    __shared__ unsigned excl[256];
    int t = threadIdx.x;
    unsigned s = 0;
    for (int k = 0; k < 32; ++k) s += cnt[t * 32 + k];
    part[t] = s;
    __syncthreads();
    if (t == 0) {
        unsigned run = 0;
        for (int q = 0; q < 256; ++q) { unsigned tmp = part[q]; excl[q] = run; run += tmp; }
        offsets[N] = run;   // == E
    }
    __syncthreads();
    unsigned run = excl[t];
    for (int k = 0; k < 32; ++k) {
        int idx = t * 32 + k;
        offsets[idx] = run;
        fillptr[idx] = run;
        run += cnt[idx];
    }
}

// ---------------- 6a: fill out-adjacency (CSR by src: lists of dst) ----------------
__global__ void k_fill(const int* __restrict__ ei, unsigned* __restrict__ fillptr,
                       unsigned* __restrict__ adj) {
    int e = blockIdx.x * blockDim.x + threadIdx.x;
    if (e < E) {
        unsigned pos = atomicAdd(&fillptr[ei[e]], 1u);
        adj[pos] = (unsigned)ei[E + e];
    }
}

// ---------------- 6b: fill in-adjacency (CSR by dst: lists of src) -----------------
__global__ void k_fill_in(const int* __restrict__ ei, unsigned* __restrict__ fillptr,
                          unsigned* __restrict__ adjin) {
    int e = blockIdx.x * blockDim.x + threadIdx.x;
    if (e < E) {
        unsigned pos = atomicAdd(&fillptr[ei[E + e]], 1u);
        adjin[pos] = (unsigned)ei[e];
    }
}

// ---------------- 7: dinv1 = 1/sqrt(1 + in-degree) ----------------
__global__ void k_dinv1(const unsigned* __restrict__ degcnt, float* __restrict__ dinv1) {
    int i = blockIdx.x * blockDim.x + threadIdx.x;
    if (i < N) dinv1[i] = rsqrtf(1.0f + (float)degcnt[i]);
}

// ------- 8: 2-hop bitmap (column-major): edge (j,m), i in outadj(m) => bit j, col i
__global__ void k_bits(const int* __restrict__ ei, const unsigned* __restrict__ offsets,
                       const unsigned* __restrict__ adj, unsigned* __restrict__ bits) {
    int e = blockIdx.x * blockDim.x + threadIdx.x;
    if (e < E) {
        int j = ei[e];
        int m = ei[E + e];
        unsigned w = 1u << (j & 31);
        unsigned wi = (unsigned)(j >> 5);
        unsigned p0 = offsets[m], p1 = offsets[m + 1];
        for (unsigned p = p0; p < p1; ++p) {
            unsigned i2 = adj[p];
            atomicOr(&bits[(size_t)i2 * 256 + wi], w);
        }
    }
}

// ---------------- 9: column popcount -> dinv2 = 1/sqrt(popc + 1) ----------------
__global__ __launch_bounds__(256)
void k_deg2(const unsigned* __restrict__ bits, float* __restrict__ dinv2) {
    __shared__ unsigned red[256];
    int i = blockIdx.x;
    int t = threadIdx.x;
    red[t] = __popc(bits[(size_t)i * 256 + t]);
    __syncthreads();
    for (int s = 128; s > 0; s >>= 1) {
        if (t < s) red[t] += red[t + s];
        __syncthreads();
    }
    if (t == 0) dinv2[i] = rsqrtf(1.0f + (float)red[0]);
}

// ---------------- 10: g1 = dinv1*h1, g2 = dinv2*h2 ----------------
__global__ void k_g(const float* __restrict__ h, const float* __restrict__ dinv1,
                    const float* __restrict__ dinv2, float* __restrict__ g1,
                    float* __restrict__ g2) {
    int i = blockIdx.x, t = threadIdx.x;   // 8192 x 128
    g1[(size_t)i * DOUT + t] = dinv1[i] * h[(size_t)i * HN + t];
    g2[(size_t)i * DOUT + t] = dinv2[i] * h[(size_t)i * HN + DOUT + t];
}

// ---------------- 11: bitmap SpMM + finalize part-2 output columns ----------------
__global__ __launch_bounds__(128)
void k_spmm2(const unsigned* __restrict__ bits, const float* __restrict__ g2,
             const float* __restrict__ dinv2, const float* __restrict__ b2,
             float* __restrict__ out) {
    int i = blockIdx.x, t = threadIdx.x;
    float a = g2[(size_t)i * DOUT + t];         // appended self-loop (+1 on diagonal)
    const unsigned* col = bits + (size_t)i * 256;
    for (int w = 0; w < 256; ++w) {
        unsigned word = col[w];                 // uniform across the block
        while (word) {
            int b = __builtin_ctz(word);
            word &= word - 1;
            a += g2[(size_t)(w * 32 + b) * DOUT + t];
        }
    }
    out[(size_t)i * HN + DOUT + t] = ALPHA * (dinv2[i] * a + b2[t]);
}

// ---------------- 12: part-1 atomic-free gather over in-CSR + finalize -------------
// out1[i] = (1-a) * ( dinv1[i] * ( g1[i] + sum_{(s->i)} g1[s] ) + b1 )
__global__ __launch_bounds__(128)
void k_part1(const unsigned* __restrict__ offsIn, const unsigned* __restrict__ adjin,
             const float* __restrict__ g1, const float* __restrict__ dinv1,
             const float* __restrict__ b1, float* __restrict__ out) {
    int i = blockIdx.x, t = threadIdx.x;
    float acc = g1[(size_t)i * DOUT + t];       // self loop: dinv1^2 * h1 after final scale
    unsigned p0 = offsIn[i], p1 = offsIn[i + 1];
    for (unsigned p = p0; p < p1; ++p) {
        unsigned s = adjin[p];
        acc += g1[(size_t)s * DOUT + t];
    }
    out[(size_t)i * HN + t] = (1.0f - ALPHA) * (dinv1[i] * acc + b1[t]);
}

// ---------------- host launcher ----------------
static inline size_t align256(size_t v) { return (v + 255) & ~(size_t)255; }

extern "C" void kernel_launch(void* const* d_in, const int* in_sizes, int n_in,
                              void* d_out, int out_size, void* d_ws, size_t ws_size,
                              hipStream_t stream) {
    const float* x  = (const float*)d_in[0];
    const int*   ei = (const int*)d_in[1];     // [2, E] flattened: src then dst
    const float* W1 = (const float*)d_in[2];
    const float* b1 = (const float*)d_in[3];
    const float* W2 = (const float*)d_in[4];
    const float* b2 = (const float*)d_in[5];
    float* out = (float*)d_out;

    // carve workspace (~42 MiB)
    char* p = (char*)d_ws;
    size_t off = 0;
    auto carve = [&](size_t bytes) { void* r = p + off; off = align256(off + bytes); return r; };
    __bf16*  xhi    = (__bf16*)carve((size_t)N * DIN * 2);
    __bf16*  xlo    = (__bf16*)carve((size_t)N * DIN * 2);
    __bf16*  whi    = (__bf16*)carve((size_t)DIN * HN * 2);
    __bf16*  wlo    = (__bf16*)carve((size_t)DIN * HN * 2);
    float*   h      = (float*)carve((size_t)N * HN * 4);
    unsigned* degcnt  = (unsigned*)carve((size_t)N * 4);
    unsigned* counts  = (unsigned*)carve((size_t)N * 4);
    unsigned* offsets = (unsigned*)carve((size_t)(N + 1) * 4);
    unsigned* fillptr = (unsigned*)carve((size_t)N * 4);
    unsigned* offsIn  = (unsigned*)carve((size_t)(N + 1) * 4);
    unsigned* fillIn  = (unsigned*)carve((size_t)N * 4);
    unsigned* adj     = (unsigned*)carve((size_t)E * 4);
    unsigned* adjin   = (unsigned*)carve((size_t)E * 4);
    unsigned* bits    = (unsigned*)carve((size_t)N * 256 * 4);
    float*   dinv1  = (float*)carve((size_t)N * 4);
    float*   dinv2  = (float*)carve((size_t)N * 4);
    float*   g1     = (float*)carve((size_t)N * DOUT * 4);
    float*   g2     = (float*)carve((size_t)N * DOUT * 4);

    (void)hipMemsetAsync(bits, 0, (size_t)N * 256 * 4, stream);
    k_init<<<N / 256, 256, 0, stream>>>(degcnt, counts);
    k_conv_x<<<(N * DIN) / 256, 256, 0, stream>>>(x, xhi, xlo);
    k_conv_w<<<(16 * 16 * 32 * 16) / 256, 256, 0, stream>>>(W1, W2, whi, wlo);
    k_gemm<<<1024, 256, 0, stream>>>(xhi, xlo, whi, wlo, h);
    k_edge_count<<<E / 256, 256, 0, stream>>>(ei, counts, degcnt);
    k_scan<<<1, 256, 0, stream>>>(counts, offsets, fillptr);
    k_scan<<<1, 256, 0, stream>>>(degcnt, offsIn, fillIn);
    k_fill<<<E / 256, 256, 0, stream>>>(ei, fillptr, adj);
    k_fill_in<<<E / 256, 256, 0, stream>>>(ei, fillIn, adjin);
    k_dinv1<<<N / 256, 256, 0, stream>>>(degcnt, dinv1);
    k_bits<<<E / 256, 256, 0, stream>>>(ei, offsets, adj, bits);
    k_deg2<<<N, 256, 0, stream>>>(bits, dinv2);
    k_g<<<N, DOUT, 0, stream>>>(h, dinv1, dinv2, g1, g2);
    k_spmm2<<<N, DOUT, 0, stream>>>(bits, g2, dinv2, b2, out);
    k_part1<<<N, DOUT, 0, stream>>>(offsIn, adjin, g1, dinv1, b1, out);
}
